// Semantic_Group_7876970021379
// MI455X (gfx1250) — matrix operand order repaired and checked by reading, not compiled, not run on hardware
//
#include <hip/hip_runtime.h>
#include <hip/hip_bf16.h>

// ---------------------------------------------------------------------------
// EdgeConv block for MI455X (gfx1250, wave32, WMMA).
//   B=4, N=4096, C=768, K=4, O1=256.
// Pipeline:
//   1) k_prep_h : h=x+pos -> bf16, sq=||h||^2            (bandwidth, ~7us)
//   2) k_prep_w : bf16 weights + folded BN scale/bias
//   3) k_knn    : d = 2 h.h^T - sq_i - sq_j via bf16 WMMA, top-4 per row
//   4) k_uv     : v = h.w1a^T, u = h.(w1b-w1a)^T via WMMA (edge-MLP algebra)
//   5) k_gather : max_k v[idx_k] + u, BN1+lrelu -> bf16
//   6) k_out    : z = y.w2^T via WMMA, BN2+lrelu -> f32 out
// ---------------------------------------------------------------------------

#define B_   4
#define N_   4096
#define C_   768
#define BN_  (B_ * N_)
#define O1_  256

typedef __attribute__((ext_vector_type(16))) __bf16 v16bf;
typedef __attribute__((ext_vector_type(8)))  float  v8f;

union V16U { v16bf v; uint4 q[2]; };

static __device__ inline v16bf ld_pair(const void* p0, const void* p1) {
  V16U t;
  t.q[0] = *(const uint4*)p0;    // b128
  t.q[1] = *(const uint4*)p1;    // b128
  return t.v;
}

static __device__ inline v8f wmma_bf16(v16bf a, v16bf b, v8f c) {
  return __builtin_amdgcn_wmma_f32_16x16x32_bf16(false, a, false, b, (short)0, c,
                                                 false, false);
}

struct Top4 {
  float v0, v1, v2, v3;
  int   i0, i1, i2, i3;
  __device__ void init() {
    v0 = v1 = v2 = v3 = -3.4e38f;
    i0 = i1 = i2 = i3 = 0;
  }
  __device__ void ins(float v, int id) {
    if (v <= v3) return;
    if (v > v0) {
      v3 = v2; i3 = i2; v2 = v1; i2 = i1; v1 = v0; i1 = i0; v0 = v; i0 = id;
    } else if (v > v1) {
      v3 = v2; i3 = i2; v2 = v1; i2 = i1; v1 = v; i1 = id;
    } else if (v > v2) {
      v3 = v2; i3 = i2; v2 = v; i2 = id;
    } else {
      v3 = v; i3 = id;
    }
  }
};

// ---------------------------------------------------------------- kernel 1 --
__global__ void k_prep_h(const float* __restrict__ x, const float* __restrict__ pos,
                         __bf16* __restrict__ hb, float* __restrict__ sq) {
  const int bn  = blockIdx.x;
  const int tid = threadIdx.x;
  const size_t base = (size_t)bn * C_;
  __shared__ float red[256];
  float acc = 0.f;
  for (int c = tid; c < C_; c += 256) {
    float v = x[base + c] + pos[base + c];
    hb[base + c] = (__bf16)v;
    acc += v * v;
  }
  red[tid] = acc;
  __syncthreads();
  for (int s = 128; s > 0; s >>= 1) {
    if (tid < s) red[tid] += red[tid + s];
    __syncthreads();
  }
  if (tid == 0) sq[bn] = red[0];
}

// ---------------------------------------------------------------- kernel 2 --
__global__ void k_prep_w(const float* __restrict__ w1, const float* __restrict__ w2,
                         const float* g1, const float* be1, const float* m1,
                         const float* va1, const float* g2, const float* be2,
                         const float* m2, const float* va2,
                         __bf16* __restrict__ w1a, __bf16* __restrict__ wd,
                         __bf16* __restrict__ w2b,
                         float* s1, float* bb1, float* s2, float* bb2) {
  const int i = blockIdx.x * 256 + threadIdx.x;   // grid covers 196608
  if (i < O1_ * C_) {
    const int o = i / C_, c = i - o * C_;
    const float a = w1[(size_t)o * (2 * C_) + c];
    const float b = w1[(size_t)o * (2 * C_) + C_ + c];
    w1a[i] = (__bf16)a;
    wd[i]  = (__bf16)(b - a);
  }
  if (i < C_ * O1_) w2b[i] = (__bf16)w2[i];
  if (i < O1_) {
    const float s = g1[i] * rsqrtf(va1[i] + 1e-5f);
    s1[i] = s; bb1[i] = be1[i] - m1[i] * s;
  }
  if (i < C_) {
    const float s = g2[i] * rsqrtf(va2[i] + 1e-5f);
    s2[i] = s; bb2[i] = be2[i] - m2[i] * s;
  }
}

// ---------------------------------------------------------------- kernel 3 --
// One block = one batch x 16-row tile. 8 waves split the 4096 columns.
__global__ void __launch_bounds__(256)
k_knn(const __bf16* __restrict__ hb, const float* __restrict__ sq,
      int* __restrict__ knn_idx) {
  const int tid  = threadIdx.x;
  const int wave = tid >> 5, lane = tid & 31;
  const int half = lane >> 4, l15 = lane & 15;
  const int b       = blockIdx.x >> 8;
  const int rowbase = (blockIdx.x & 255) << 4;

  __shared__ __align__(16) __bf16 As[16 * C_];   // 24 KB A tile (row-major)
  __shared__ float Dt[8][16][17];                // per-wave 16x16 d tile
  __shared__ float Mv[8][16][4];
  __shared__ int   Mi[8][16][4];
  __shared__ float SqRow[16];

  const __bf16* hbatch = hb + (size_t)b * N_ * C_;
  {
    const uint4* src = (const uint4*)(hbatch + (size_t)rowbase * C_);
    uint4* dst = (uint4*)As;
    for (int t = tid; t < (16 * C_) / 8; t += 256) dst[t] = src[t];
    if (tid < 16) SqRow[tid] = sq[(size_t)b * N_ + rowbase + tid];
  }
  __syncthreads();

  Top4 t4; t4.init();
  const float*  sqb  = sq + (size_t)b * N_;
  const __bf16* aptr = As + (size_t)l15 * C_ + half * 16;   // A 16x32 lane layout

  for (int ct = wave; ct < N_ / 16; ct += 8) {
    const int colbase = ct << 4;
    const __bf16* bptr = hbatch + (size_t)(colbase + l15) * C_ + half * 16;
    __builtin_prefetch(hbatch + (size_t)(colbase + 128 + l15) * C_, 0, 1);
    v8f acc = {};
    for (int kk = 0; kk < C_; kk += 32) {
      v16bf a  = ld_pair(aptr + kk, aptr + kk + 8);    // ds_load_b128 x2
      v16bf bm = ld_pair(bptr + kk, bptr + kk + 8);    // global b128 x2
      acc = wmma_bf16(a, bm, acc);
    }
    const float sqc = sqb[colbase + l15];
#pragma unroll
    for (int r = 0; r < 8; ++r) {
      const int row = r + 8 * half;                    // C-frag: M = r + 8*(lane>=16)
      Dt[wave][row][l15] = 2.f * acc[r] - SqRow[row] - sqc;
    }
    asm volatile("s_wait_dscnt 0x0" ::: "memory");     // cross-lane LDS RAW (same wave)
    if (lane < 16) {
#pragma unroll
      for (int c2 = 0; c2 < 16; ++c2) t4.ins(Dt[wave][lane][c2], colbase + c2);
    }
  }

  if (lane < 16) {
    Mv[wave][lane][0] = t4.v0; Mi[wave][lane][0] = t4.i0;
    Mv[wave][lane][1] = t4.v1; Mi[wave][lane][1] = t4.i1;
    Mv[wave][lane][2] = t4.v2; Mi[wave][lane][2] = t4.i2;
    Mv[wave][lane][3] = t4.v3; Mi[wave][lane][3] = t4.i3;
  }
  __syncthreads();
  if (wave == 0 && lane < 16) {
    Top4 f; f.init();
#pragma unroll
    for (int w = 0; w < 8; ++w)
#pragma unroll
      for (int k = 0; k < 4; ++k) f.ins(Mv[w][lane][k], Mi[w][lane][k]);
    int* op = knn_idx + ((size_t)b * N_ + rowbase + lane) * 4;
    op[0] = f.i0; op[1] = f.i1; op[2] = f.i2; op[3] = f.i3;
  }
}

// ---------------------------------------------------------------- kernel 4 --
// v = h . w1a^T, u = h . (w1b - w1a)^T  (shared A, two accumulators)
__global__ void __launch_bounds__(256)
k_uv(const __bf16* __restrict__ hb, const __bf16* __restrict__ w1a,
     const __bf16* __restrict__ wd, float* __restrict__ u, float* __restrict__ v) {
  const int tid  = threadIdx.x;
  const int wave = tid >> 5, lane = tid & 31;
  const int half = lane >> 4, l15 = lane & 15;
  const int rowbase = blockIdx.x << 4;

  __shared__ __align__(16) __bf16 As[16 * C_];
  {
    const uint4* src = (const uint4*)(hb + (size_t)rowbase * C_);
    uint4* dst = (uint4*)As;
    for (int t = tid; t < (16 * C_) / 8; t += 256) dst[t] = src[t];
  }
  __syncthreads();

  const __bf16* aptr = As + (size_t)l15 * C_ + half * 16;
  for (int j = 0; j < 2; ++j) {
    const int obase = (wave * 2 + j) << 4;
    const __bf16* wa  = w1a + (size_t)(obase + l15) * C_ + half * 16;
    const __bf16* wdp = wd  + (size_t)(obase + l15) * C_ + half * 16;
    v8f vacc = {}, uacc = {};
    for (int kk = 0; kk < C_; kk += 32) {
      v16bf a  = ld_pair(aptr + kk, aptr + kk + 8);
      v16bf b1 = ld_pair(wa + kk,  wa + kk + 8);
      v16bf b2 = ld_pair(wdp + kk, wdp + kk + 8);
      vacc = wmma_bf16(a, b1, vacc);
      uacc = wmma_bf16(a, b2, uacc);
    }
#pragma unroll
    for (int r = 0; r < 8; ++r) {
      const int row = rowbase + r + 8 * half;
      v[(size_t)row * O1_ + obase + l15] = vacc[r];
      u[(size_t)row * O1_ + obase + l15] = uacc[r];
    }
  }
}

// ---------------------------------------------------------------- kernel 5 --
__global__ void k_gather(const float* __restrict__ v, const float* __restrict__ u,
                         const int* __restrict__ knn_idx,
                         const float* __restrict__ s1, const float* __restrict__ bb1,
                         __bf16* __restrict__ ybf) {
  const int bn = blockIdx.x;
  const int o  = threadIdx.x;
  const int b  = bn >> 12;                         // bn / N_
  const int4 id = *(const int4*)(knn_idx + (size_t)bn * 4);
  const size_t nb = (size_t)b * N_;
  const float m0 = v[(nb + id.x) * (size_t)O1_ + o];
  const float m1 = v[(nb + id.y) * (size_t)O1_ + o];
  const float m2 = v[(nb + id.z) * (size_t)O1_ + o];
  const float m3 = v[(nb + id.w) * (size_t)O1_ + o];
  const float m  = fmaxf(fmaxf(m0, m1), fmaxf(m2, m3));
  float t = s1[o] * (m + u[(size_t)bn * O1_ + o]) + bb1[o];
  t = (t >= 0.f) ? t : 0.2f * t;
  ybf[(size_t)bn * O1_ + o] = (__bf16)t;
}

// ---------------------------------------------------------------- kernel 6 --
__global__ void __launch_bounds__(256)
k_out(const __bf16* __restrict__ ybf, const __bf16* __restrict__ w2b,
      const float* __restrict__ s2, const float* __restrict__ bb2,
      float* __restrict__ out) {
  const int tid  = threadIdx.x;
  const int wave = tid >> 5, lane = tid & 31;
  const int half = lane >> 4, l15 = lane & 15;
  const int rowbase = blockIdx.x << 4;

  __shared__ __align__(16) __bf16 As[16 * O1_];    // 8 KB
  {
    const uint4* src = (const uint4*)(ybf + (size_t)rowbase * O1_);
    uint4* dst = (uint4*)As;
    for (int t = tid; t < (16 * O1_) / 8; t += 256) dst[t] = src[t];
  }
  __syncthreads();

  const __bf16* aptr = As + (size_t)l15 * O1_ + half * 16;
  for (int j = 0; j < 6; ++j) {
    const int obase = (wave * 6 + j) << 4;
    const __bf16* wp = w2b + (size_t)(obase + l15) * O1_ + half * 16;
    v8f acc = {};
    for (int kk = 0; kk < O1_; kk += 32) {
      v16bf a  = ld_pair(aptr + kk, aptr + kk + 8);
      v16bf bm = ld_pair(wp + kk, wp + kk + 8);
      acc = wmma_bf16(a, bm, acc);
    }
    const int col = obase + l15;
    const float sc = s2[col], bi = bb2[col];
#pragma unroll
    for (int r = 0; r < 8; ++r) {
      const int row = rowbase + r + 8 * half;
      float t = sc * acc[r] + bi;
      t = (t >= 0.f) ? t : 0.2f * t;
      out[(size_t)row * C_ + col] = t;
    }
  }
}

// ------------------------------------------------------------------ launch --
extern "C" void kernel_launch(void* const* d_in, const int* in_sizes, int n_in,
                              void* d_out, int out_size, void* d_ws, size_t ws_size,
                              hipStream_t stream) {
  const float* x      = (const float*)d_in[0];
  const float* pos    = (const float*)d_in[1];
  const float* w1     = (const float*)d_in[2];
  const float* gamma1 = (const float*)d_in[3];
  const float* beta1  = (const float*)d_in[4];
  const float* mean1  = (const float*)d_in[5];
  const float* var1   = (const float*)d_in[6];
  const float* w2     = (const float*)d_in[7];
  const float* gamma2 = (const float*)d_in[8];
  const float* beta2  = (const float*)d_in[9];
  const float* mean2  = (const float*)d_in[10];
  const float* var2   = (const float*)d_in[11];
  float* out = (float*)d_out;

  char* p = (char*)d_ws;
  auto alloc = [&](size_t bytes) {
    char* r = p;
    p += (bytes + 255) & ~(size_t)255;
    return (void*)r;
  };
  __bf16* hb   = (__bf16*)alloc((size_t)BN_ * C_ * 2);     // 25.2 MB
  float*  sq   = (float*) alloc((size_t)BN_ * 4);
  int*    idx  = (int*)   alloc((size_t)BN_ * 4 * 4);
  float*  u    = (float*) alloc((size_t)BN_ * O1_ * 4);    // 16.8 MB
  float*  v    = (float*) alloc((size_t)BN_ * O1_ * 4);    // 16.8 MB
  __bf16* ybf  = (__bf16*)alloc((size_t)BN_ * O1_ * 2);    //  8.4 MB
  __bf16* w1a  = (__bf16*)alloc((size_t)O1_ * C_ * 2);
  __bf16* wd   = (__bf16*)alloc((size_t)O1_ * C_ * 2);
  __bf16* w2b  = (__bf16*)alloc((size_t)C_ * O1_ * 2);
  float*  s1   = (float*) alloc((size_t)O1_ * 4);
  float*  bb1  = (float*) alloc((size_t)O1_ * 4);
  float*  s2   = (float*) alloc((size_t)C_ * 4);
  float*  bb2  = (float*) alloc((size_t)C_ * 4);

  k_prep_h<<<BN_, 256, 0, stream>>>(x, pos, hb, sq);
  k_prep_w<<<(O1_ * C_ + 255) / 256, 256, 0, stream>>>(
      w1, w2, gamma1, beta1, mean1, var1, gamma2, beta2, mean2, var2,
      w1a, wd, w2b, s1, bb1, s2, bb2);
  k_knn<<<B_ * (N_ / 16), 256, 0, stream>>>(hb, sq, idx);
  k_uv<<<BN_ / 16, 256, 0, stream>>>(hb, w1a, wd, u, v);
  k_gather<<<BN_, 256, 0, stream>>>(v, u, idx, s1, bb1, ybf);
  k_out<<<BN_ / 16, 256, 0, stream>>>(ybf, w2b, s2, bb2, out);
}
